// TimeMIL_58411555225897
// MI455X (gfx1250) — matrix operand order repaired and checked by reading, not compiled, not run on hardware
//
#include <hip/hip_runtime.h>

// ============================================================================
// TimeMIL forward for MI455X (gfx1250, wave32, WMMA).
// All heavy math runs through v_wmma_f32_16x16x32_f16 (f16 in, f32 accum).
// All GEMM shapes in this net have M,N %64==0 and K %32==0 -> no bounds checks.
// Strided-gather strides (ldb, KW, L) are template/compile-time constants so
// element loads become immediate-offset clauses from one base pointer.
// Requires ws_size >= ~430 MB.
// ============================================================================

typedef __attribute__((ext_vector_type(16))) _Float16 v16h;
typedef __attribute__((ext_vector_type(8)))  float    v8f;

// ---------------------------------------------------------------------------
// Workspace layout (bytes). Regions P0/P1/P2 are reused across phases.
// ---------------------------------------------------------------------------
static constexpr size_t SZ_TOK = (size_t)8 * 1026 * 512 * 4;   // token stream f32
static constexpr size_t SZ_BIG = (size_t)64 * 1280 * 256 * 4;  // a1/a3/ap
static constexpr size_t SZ_PV  = (size_t)64 * 1280 * 64 * 4;   // q/k/v per-head
static constexpr size_t SZ_LM  = (size_t)64 * 256 * 64 * 4;    // landmarks
static constexpr size_t SZ_SQ  = (size_t)64 * 256 * 256 * 4;   // 256x256 per head
static constexpr size_t SZ_A16 = (size_t)8 * 512 * 1024 * 2;   // f16 activation

static constexpr size_t O_TOK  = 0;
static constexpr size_t O_P0   = SZ_TOK;            // inception arena / A1
static constexpr size_t O_P1   = O_P0 + SZ_BIG;     // qkv_raw -> A3 -> aout+merg+proj
static constexpr size_t O_P2   = O_P1 + SZ_BIG;     // pos -> lnx+xpad -> AP
static constexpr size_t O_Q    = O_P2 + SZ_BIG;
static constexpr size_t O_K    = O_Q + SZ_PV;
static constexpr size_t O_V    = O_K + SZ_PV;
static constexpr size_t O_QL   = O_V + SZ_PV;
static constexpr size_t O_KL   = O_QL + SZ_LM;
static constexpr size_t O_A2   = O_KL + SZ_LM;
static constexpr size_t O_Z    = O_A2 + SZ_SQ;
static constexpr size_t O_ZB   = O_Z + SZ_SQ;
static constexpr size_t O_T1   = O_ZB + SZ_SQ;
static constexpr size_t O_T2   = O_T1 + SZ_SQ;
static constexpr size_t O_A3V  = O_T2 + SZ_SQ;
static constexpr size_t O_RC   = O_A3V + SZ_LM;
static constexpr size_t O_DEN  = O_RC + 512;
// phase A (inception) inside P0:
static constexpr size_t O_XA   = O_P0;
static constexpr size_t O_XB   = O_XA + SZ_A16;
static constexpr size_t O_XBOT = O_XB + SZ_A16;
static constexpr size_t O_MP   = O_XBOT + (size_t)8 * 128 * 1024 * 2;
static constexpr size_t O_S    = O_MP + SZ_A16;
static constexpr size_t O_RES  = O_S + SZ_A16;
// inside P1:
static constexpr size_t O_QKVRAW = O_P1;
static constexpr size_t O_A3     = O_P1;
static constexpr size_t O_AOUT   = O_P1;
static constexpr size_t O_MERG   = O_P1 + SZ_PV;
static constexpr size_t O_PROJ   = O_P1 + 2 * SZ_PV;
// inside P2:
static constexpr size_t O_POS  = O_P2;
static constexpr size_t O_LNX  = O_P2;
static constexpr size_t O_XPAD = O_P2 + SZ_TOK;
static constexpr size_t O_AP   = O_P2;

// A fragment: 16 contiguous floats per lane (k = 16*lh + s) -> 4x float4 loads.
__device__ __forceinline__ v16h load_a_frag(const float* __restrict__ row, int lh) {
  const float4* p0 = reinterpret_cast<const float4*>(row + lh * 16);
  const float4* p1 = reinterpret_cast<const float4*>(row + lh * 16 + 8);
  float4 f0 = p0[0], f1 = p0[1], f2 = p1[0], f3 = p1[1];
  v16h a;
  a[0] = (_Float16)f0.x;  a[1] = (_Float16)f0.y;  a[2] = (_Float16)f0.z;  a[3] = (_Float16)f0.w;
  a[4] = (_Float16)f1.x;  a[5] = (_Float16)f1.y;  a[6] = (_Float16)f1.z;  a[7] = (_Float16)f1.w;
  a[8] = (_Float16)f2.x;  a[9] = (_Float16)f2.y;  a[10] = (_Float16)f2.z; a[11] = (_Float16)f2.w;
  a[12] = (_Float16)f3.x; a[13] = (_Float16)f3.y; a[14] = (_Float16)f3.z; a[15] = (_Float16)f3.w;
  return a;
}
// B^T fragment: 16 contiguous floats per lane -> 4x float4 loads.
__device__ __forceinline__ v16h load_bt_frag(const float* __restrict__ row) {
  const float4* p = reinterpret_cast<const float4*>(row);
  float4 f0 = p[0], f1 = p[1], f2 = p[2], f3 = p[3];
  v16h b;
  b[0] = (_Float16)f0.x;  b[1] = (_Float16)f0.y;  b[2] = (_Float16)f0.z;  b[3] = (_Float16)f0.w;
  b[4] = (_Float16)f1.x;  b[5] = (_Float16)f1.y;  b[6] = (_Float16)f1.z;  b[7] = (_Float16)f1.w;
  b[8] = (_Float16)f2.x;  b[9] = (_Float16)f2.y;  b[10] = (_Float16)f2.z; b[11] = (_Float16)f2.w;
  b[12] = (_Float16)f3.x; b[13] = (_Float16)f3.y; b[14] = (_Float16)f3.z; b[15] = (_Float16)f3.w;
  return b;
}

// ---------------------------------------------------------------------------
// Batched GEMM: C = alpha * f16(A) @ f16(B [or B^T]) + bias + beta*C
// block = 128 threads (4 waves, 2x2); wave tile 32x32 (2x2 WMMA fragments);
// block tile 64x64. Requires M%64==0, N%64==0, K%32==0 (true for all calls).
// LDB compile-time -> column gathers are immediate-offset loads off one base.
// grid: (N/64, M/64, batch)
// ---------------------------------------------------------------------------
template <int BT, int LDB>
__global__ void gemm_f32_wmma(const float* __restrict__ A, const float* __restrict__ B,
                              float* __restrict__ C,
                              int K, long sA, long sB, long sC,
                              int lda, int ldc,
                              const float* __restrict__ bias,
                              float alpha, float beta) {
  A += (long)blockIdx.z * sA;
  B += (long)blockIdx.z * sB;
  C += (long)blockIdx.z * sC;
  int lane = threadIdx.x & 31;
  int wave = threadIdx.x >> 5;
  int m0 = blockIdx.y * 64 + (wave >> 1) * 32;
  int n0 = blockIdx.x * 64 + (wave & 1) * 32;
  int lm = lane & 15, lh = lane >> 4;
  v8f acc00 = {}, acc01 = {}, acc10 = {}, acc11 = {};
  const float* arow0 = A + (long)(m0 + lm) * lda;
  const float* arow1 = arow0 + (long)16 * lda;
  const float* bbase = BT ? (B + (long)(n0 + lm) * LDB + lh * 16)
                          : (B + (long)(lh * 16) * LDB + n0 + lm);
  for (int k0 = 0; k0 < K; k0 += 32) {
    v16h a0 = load_a_frag(arow0 + k0, lh);
    v16h a1 = load_a_frag(arow1 + k0, lh);
    v16h b0, b1;
    if (BT) {
      b0 = load_bt_frag(bbase);
      b1 = load_bt_frag(bbase + 16 * LDB);
      bbase += 32;
    } else {
#pragma unroll
      for (int i = 0; i < 16; ++i) {     // immediate offsets: i*LDB, i*LDB+16
        b0[i] = (_Float16)bbase[i * LDB];
        b1[i] = (_Float16)bbase[i * LDB + 16];
      }
      bbase += 32 * LDB;
    }
    acc00 = __builtin_amdgcn_wmma_f32_16x16x32_f16(false, a0, false, b0, (short)0, acc00, false, false);
    acc01 = __builtin_amdgcn_wmma_f32_16x16x32_f16(false, a0, false, b1, (short)0, acc01, false, false);
    acc10 = __builtin_amdgcn_wmma_f32_16x16x32_f16(false, a1, false, b0, (short)0, acc10, false, false);
    acc11 = __builtin_amdgcn_wmma_f32_16x16x32_f16(false, a1, false, b1, (short)0, acc11, false, false);
  }
  v8f* accs[2][2] = {{&acc00, &acc01}, {&acc10, &acc11}};
#pragma unroll
  for (int mi = 0; mi < 2; ++mi) {
#pragma unroll
    for (int ni = 0; ni < 2; ++ni) {
      v8f acc = *accs[mi][ni];
      int n = n0 + ni * 16 + lm;
      float bs = bias ? bias[n] : 0.f;
#pragma unroll
      for (int r = 0; r < 8; ++r) {
        int m = m0 + mi * 16 + lh * 8 + r;
        long idx = (long)m * ldc + n;
        float v = alpha * acc[r] + bs;
        if (beta != 0.f) v += beta * C[idx];
        C[idx] = v;
      }
    }
  }
}

// ---------------------------------------------------------------------------
// Direct conv1d (SAME, odd KW, L=1024) as sum-over-taps WMMA GEMMs.
// X f16 [B][CI][1024]; W f32 (CO,CI,KW); fused BN (g*rsqrt(1+1e-5), b) + ReLU
// into concat buffer Y f16 [B][CtotOut][1024] at channel offset coOff.
// KW compile-time -> weight gathers are immediate-offset loads off one base.
// block = 128 (4 waves); wave tile 32co x 16t; block tile 128co x 16t.
// grid: (64, CO/128, B)
// ---------------------------------------------------------------------------
template <int KW>
__global__ void conv1d_wmma(const _Float16* __restrict__ X, const float* __restrict__ W,
                            _Float16* __restrict__ Y,
                            int CI, int CtotOut, int coOff,
                            const float* __restrict__ g, const float* __restrict__ b,
                            int relu) {
  constexpr int L = 1024;
  int lane = threadIdx.x & 31;
  int wave = threadIdx.x >> 5;
  int co0 = blockIdx.y * 128 + wave * 32;
  int t0 = blockIdx.x * 16;
  int lm = lane & 15, lh = lane >> 4;
  constexpr int pad = (KW - 1) / 2;
  const _Float16* Xb = X + (long)blockIdx.z * CI * L;
  v8f acc0 = {}, acc1 = {};
  for (int j = 0; j < KW; ++j) {
    int t = t0 + lm + j - pad;
    bool ok = (t >= 0 && t < L);
    const _Float16* xcol = Xb + t + (long)(lh * 16) * L;
    const float* w0 = W + ((long)(co0 + lm) * CI) * KW + j + lh * 16 * KW;  // row co0+lm
    const float* w1 = w0 + (long)16 * CI * KW;                             // row +16
    for (int c0 = 0; c0 < CI; c0 += 32) {
      v16h bb;
      if (ok) {
#pragma unroll
        for (int i = 0; i < 16; ++i)
          bb[i] = xcol[i * L];                  // immediate offsets: i*2048 bytes
      } else {
#pragma unroll
        for (int i = 0; i < 16; ++i) bb[i] = (_Float16)0.f;
      }
      v16h a0, a1;
#pragma unroll
      for (int i = 0; i < 16; ++i) {            // immediate offsets: kk*KW*4 bytes
        int kk = i;
        a0[i] = (_Float16)w0[kk * KW];
        a1[i] = (_Float16)w1[kk * KW];
      }
      acc0 = __builtin_amdgcn_wmma_f32_16x16x32_f16(false, a0, false, bb, (short)0, acc0, false, false);
      acc1 = __builtin_amdgcn_wmma_f32_16x16x32_f16(false, a1, false, bb, (short)0, acc1, false, false);
      xcol += 32 * L;
      w0 += 32 * KW;
      w1 += 32 * KW;
    }
  }
  v8f* accs[2] = {&acc0, &acc1};
#pragma unroll
  for (int half = 0; half < 2; ++half) {
    v8f acc = *accs[half];
#pragma unroll
    for (int r = 0; r < 8; ++r) {
      int co = co0 + half * 16 + lh * 8 + r;
      int t = t0 + lm;
      float v = acc[r];
      if (g) {
        int c = coOff + co;
        v = v * (g[c] * rsqrtf(1.0f + 1e-5f)) + b[c];
      }
      if (relu) v = fmaxf(v, 0.f);
      Y[((long)blockIdx.z * CtotOut + coOff + co) * (long)L + t] = (_Float16)v;
    }
  }
}

// ------------------------------- elementwise -------------------------------
__global__ void k_prep_input(const float* __restrict__ x, _Float16* xa, _Float16* res,
                             long total) {
  long i = (long)blockIdx.x * 256 + threadIdx.x;
  if (i >= total) return;
  int t = (int)(i & 1023);
  long bc = i >> 10;
  int c = (int)(bc & 31);
  int b = (int)(bc >> 5);
  float v = x[(((long)b << 10) + t) * 32 + c];
  xa[i] = (_Float16)v;
  res[i] = (_Float16)v;
}

__global__ void k_maxpool3(const _Float16* __restrict__ x, _Float16* __restrict__ y,
                           long total, int L) {
  long i = (long)blockIdx.x * 256 + threadIdx.x;
  if (i >= total) return;
  int t = (int)(i & (L - 1));
  float m = (float)x[i];
  if (t > 0) m = fmaxf(m, (float)x[i - 1]);
  if (t < L - 1) m = fmaxf(m, (float)x[i + 1]);
  y[i] = (_Float16)m;
}

__global__ void k_res_add_relu(_Float16* __restrict__ h, const _Float16* __restrict__ s,
                               const _Float16* __restrict__ res,
                               const float* __restrict__ g, const float* __restrict__ b,
                               int use_bn, _Float16* __restrict__ res_out, long total) {
  long i = (long)blockIdx.x * 256 + threadIdx.x;
  if (i >= total) return;
  int c = (int)((i >> 10) & 511);
  float sv = use_bn ? ((float)res[i] * (g[c] * rsqrtf(1.0f + 1e-5f)) + b[c])
                    : (float)s[i];
  float v = fmaxf((float)h[i] + sv, 0.f);
  h[i] = (_Float16)v;
  res_out[i] = (_Float16)v;
}

__global__ void k_build_tokens(const _Float16* __restrict__ feats,
                               const float* __restrict__ cls, float* __restrict__ tok,
                               long total) {
  long i = (long)blockIdx.x * 256 + threadIdx.x;
  if (i >= total) return;
  int d = (int)(i & 511);
  long bt = i >> 9;
  int it = (int)(bt % 1026);
  int b = (int)(bt / 1026);
  tok[i] = (it < 2) ? cls[it * 512 + d]
                    : (float)feats[((long)b * 512 + d) * 1024 + (it - 2)];
}

__global__ void k_wavelet_pos(const float* __restrict__ tok, const float* __restrict__ w1,
                              const float* __restrict__ w2, const float* __restrict__ w3,
                              float* __restrict__ pos, long total) {
  long i = (long)blockIdx.x * 256 + threadIdx.x;
  if (i >= total) return;
  int d = (int)(i & 511);
  long bt = i >> 9;
  int t = (int)(bt & 1023);
  int b = (int)(bt >> 10);
  const float* wv[3] = {w1, w2, w3};
  const float* base = tok + ((long)b * 1026 + 2) * 512 + d;
  float acc = 0.f;
#pragma unroll
  for (int w = 0; w < 3; ++w) {
    float scale = wv[w][d];
    float shift = wv[w][512 + d];
    float inv = 1.f / scale;
    float amp = 0.8673250705840776f * rsqrtf(fabsf(scale));  // 2/(sqrt(3)*pi^.25)
    for (int j = 0; j < 19; ++j) {
      int ts = t + j - 9;
      if ((unsigned)ts < 1024u) {
        float r = ((float)(j - 9) - shift) * inv;
        acc += amp * (1.f - r * r) * expf(-0.5f * r * r) * base[(long)ts * 512];
      }
    }
  }
  pos[i] = acc;
}

__global__ void k_layernorm(const float* __restrict__ x, const float* __restrict__ g,
                            const float* __restrict__ b, float* __restrict__ y) {
  long row = blockIdx.x;
  const float* xr = x + row * 512;
  float* yr = y + row * 512;
  __shared__ float s1[256], s2[256];
  int t = threadIdx.x;
  float a0 = xr[t], a1 = xr[t + 256];
  s1[t] = a0 + a1;
  s2[t] = a0 * a0 + a1 * a1;
  __syncthreads();
  for (int off = 128; off > 0; off >>= 1) {
    if (t < off) { s1[t] += s1[t + off]; s2[t] += s2[t + off]; }
    __syncthreads();
  }
  float mu = s1[0] * (1.f / 512.f);
  float var = s2[0] * (1.f / 512.f) - mu * mu;
  float r = rsqrtf(var + 1e-5f);
  yr[t] = (a0 - mu) * r * g[t] + b[t];
  yr[t + 256] = (a1 - mu) * r * g[t + 256] + b[t + 256];
}

__global__ void k_pad_tokens(const float* __restrict__ lnx, float* __restrict__ xpad,
                             long total) {
  long i = (long)blockIdx.x * 256 + threadIdx.x;
  if (i >= total) return;
  int d = (int)(i & 511);
  long bp = i >> 9;
  int p = (int)(bp % 1280);
  int b = (int)(bp / 1280);
  xpad[i] = (p < 254) ? 0.f : lnx[((long)b * 1026 + (p - 254)) * 512 + d];
}

__global__ void k_split_qkv(const float* __restrict__ raw, float* __restrict__ q,
                            float* __restrict__ k, float* __restrict__ v, long total) {
  long i = (long)blockIdx.x * 256 + threadIdx.x;
  if (i >= total) return;
  int c = (int)(i & 511);
  long bp = i >> 9;
  int p = (int)(bp % 1280);
  int b = (int)(bp / 1280);
  int h = c >> 6, d = c & 63;
  long o = (((long)(b * 8 + h)) * 1280 + p) * 64 + d;
  const float* r = raw + bp * 1536;
  q[o] = r[c] * 0.125f;  // DIM_HEAD^-0.5
  k[o] = r[512 + c];
  v[o] = r[1024 + c];
}

__global__ void k_landmarks(const float* __restrict__ q, const float* __restrict__ k,
                            float* __restrict__ ql, float* __restrict__ kl, long total) {
  long i = (long)blockIdx.x * 256 + threadIdx.x;
  if (i >= total) return;
  int d = (int)(i & 63);
  long rem = i >> 6;
  int lm = (int)(rem & 255);
  long bh = rem >> 8;
  long base = (bh * 1280 + (long)lm * 5) * 64 + d;
  float sq = 0.f, sk = 0.f;
#pragma unroll
  for (int j = 0; j < 5; ++j) { sq += q[base + j * 64]; sk += k[base + j * 64]; }
  ql[i] = sq * 0.2f;
  kl[i] = sk * 0.2f;
}

__global__ void k_softmax_rows(float* __restrict__ x, int ncols) {
  long row = blockIdx.x;
  float* xr = x + row * (long)ncols;
  __shared__ float red[256];
  int t = threadIdx.x;
  float m = -1e30f;
  for (int j = t; j < ncols; j += 256) m = fmaxf(m, xr[j]);
  red[t] = m; __syncthreads();
  for (int off = 128; off > 0; off >>= 1) {
    if (t < off) red[t] = fmaxf(red[t], red[t + off]);
    __syncthreads();
  }
  m = red[0]; __syncthreads();
  float s = 0.f;
  for (int j = t; j < ncols; j += 256) { float e = expf(xr[j] - m); xr[j] = e; s += e; }
  red[t] = s; __syncthreads();
  for (int off = 128; off > 0; off >>= 1) {
    if (t < off) red[t] += red[t + off];
    __syncthreads();
  }
  float inv = 1.f / red[0];
  for (int j = t; j < ncols; j += 256) xr[j] *= inv;
}

__global__ void k_pinv_sums(const float* __restrict__ x, float* __restrict__ rc) {
  int bh = blockIdx.x, t = threadIdx.x;
  const float* m = x + (long)bh * 65536;
  float rs = 0.f, cs = 0.f;
  for (int j = 0; j < 256; ++j) { rs += fabsf(m[t * 256 + j]); cs += fabsf(m[j * 256 + t]); }
  __shared__ float r1[256], r2[256];
  r1[t] = rs; r2[t] = cs; __syncthreads();
  for (int off = 128; off > 0; off >>= 1) {
    if (t < off) { r1[t] = fmaxf(r1[t], r1[t + off]); r2[t] = fmaxf(r2[t], r2[t + off]); }
    __syncthreads();
  }
  if (t == 0) { rc[bh * 2] = r1[0]; rc[bh * 2 + 1] = r2[0]; }
}

__global__ void k_pinv_denom(const float* __restrict__ rc, float* __restrict__ den) {
  __shared__ float r1[64], r2[64];
  int t = threadIdx.x;
  r1[t] = rc[t * 2]; r2[t] = rc[t * 2 + 1]; __syncthreads();
  for (int off = 32; off > 0; off >>= 1) {
    if (t < off) { r1[t] = fmaxf(r1[t], r1[t + off]); r2[t] = fmaxf(r2[t], r2[t + off]); }
    __syncthreads();
  }
  if (t == 0) den[0] = r1[0] * r2[0];  // global max over batch*heads (matches jnp .max())
}

__global__ void k_pinv_init(const float* __restrict__ a2, const float* __restrict__ den,
                            float* __restrict__ z, long total) {
  long i = (long)blockIdx.x * 256 + threadIdx.x;
  if (i >= total) return;
  long bh = i >> 16;
  long rem = i & 65535;
  int r = (int)(rem >> 8), c = (int)(rem & 255);
  z[i] = a2[(bh << 16) + ((long)c << 8) + r] / den[0];  // transpose / denom
}

__global__ void k_id_sub(const float* __restrict__ in, float* __restrict__ out, float c,
                         long total) {
  long i = (long)blockIdx.x * 256 + threadIdx.x;
  if (i >= total) return;
  int r = (int)((i >> 8) & 255), cc = (int)(i & 255);
  out[i] = ((r == cc) ? c : 0.f) - in[i];
}

__global__ void k_resconv(float* __restrict__ out, const float* __restrict__ v,
                          const float* __restrict__ rw, long total) {
  long i = (long)blockIdx.x * 256 + threadIdx.x;
  if (i >= total) return;
  int d = (int)(i & 63);
  long pi = i >> 6;
  int p = (int)(pi % 1280);
  long bh = pi / 1280;
  int h = (int)(bh & 7);
  float acc = out[i];
  for (int j = 0; j < 33; ++j) {
    int pp = p + j - 16;
    if ((unsigned)pp < 1280u) acc += rw[h * 33 + j] * v[(bh * 1280 + pp) * 64 + d];
  }
  out[i] = acc;
}

__global__ void k_merge(const float* __restrict__ aout, float* __restrict__ merg,
                        long total) {
  long i = (long)blockIdx.x * 256 + threadIdx.x;
  if (i >= total) return;
  int c = (int)(i & 511);
  long bp = i >> 9;
  int p = (int)(bp % 1280);
  int b = (int)(bp / 1280);
  merg[i] = aout[(((long)(b * 8 + (c >> 6))) * 1280 + p) * 64 + (c & 63)];
}

__global__ void k_add_attn(float* __restrict__ tok, const float* __restrict__ proj,
                           long total) {
  long i = (long)blockIdx.x * 256 + threadIdx.x;
  if (i >= total) return;
  int d = (int)(i & 511);
  long bi = i >> 9;
  int it = (int)(bi % 1026);
  int b = (int)(bi / 1026);
  tok[i] += proj[((long)b * 1280 + 254 + it) * 512 + d];  // out[:, -n:]
}

__global__ void k_final(const float* __restrict__ tok, float* __restrict__ out) {
  int o = blockIdx.x;
  const float* row = tok + ((long)((o >> 1) * 1026 + (o & 1))) * 512;
  __shared__ float red[256];
  int t = threadIdx.x;
  red[t] = row[t] + row[t + 256];
  __syncthreads();
  for (int off = 128; off > 0; off >>= 1) {
    if (t < off) red[t] += red[t + off];
    __syncthreads();
  }
  if (t == 0) out[o] = red[0] * (1.f / 512.f);
}

// ---------------------------------------------------------------------------
extern "C" void kernel_launch(void* const* d_in, const int* in_sizes, int n_in,
                              void* d_out, int out_size, void* d_ws, size_t ws_size,
                              hipStream_t stream) {
  (void)in_sizes; (void)n_in; (void)out_size; (void)ws_size;
  // Param flattening order (setup_inputs dict order; sc1.conv==None skipped):
  // 0:x ; modules d=0..5 at base=1+7d: +0 bottleneck,+1..3 convs(39/19/9),
  // +4 mp_conv,+5 bn_g,+6 bn_b ; 43 sc0.conv,44 sc0.g,45 sc0.b,46 sc1.g,47 sc1.b ;
  // 48 cls ; 49-51 wave1..3 ; 52-54 wave1_..3_ ; 55 pe1_w,56 pe1_b,57 pe2_w,58 pe2_b ;
  // 59-62 ln1_g,ln1_b,ln2_g,ln2_b ; 63-66 attn1{qkv,out_w,out_b,res_w} ; 67-70 attn2.
  const float* X = (const float*)d_in[0];
  auto P = [&](int i) -> const float* { return (const float*)d_in[i]; };
  char* ws = (char*)d_ws;
  auto F = [&](size_t o) -> float* { return (float*)(ws + o); };
  auto Hp = [&](size_t o) -> _Float16* { return (_Float16*)(ws + o); };
  auto ew = [](long n) -> unsigned { return (unsigned)((n + 255) / 256); };

  auto gemm = [&](const float* A, const float* Bm, float* C, int M, int N, int K,
                  long sA, long sB, long sC, int lda, int ldb, int ldc, int bT,
                  const float* bias, float alpha, float beta, int batch) {
    dim3 grid(N / 64, M / 64, batch);
    if (bT)  // all transposed-B calls have ldb == 64
      gemm_f32_wmma<1, 64><<<grid, 128, 0, stream>>>(A, Bm, C, K, sA, sB, sC, lda, ldc,
                                                     bias, alpha, beta);
    else if (ldb == 64)
      gemm_f32_wmma<0, 64><<<grid, 128, 0, stream>>>(A, Bm, C, K, sA, sB, sC, lda, ldc,
                                                     bias, alpha, beta);
    else if (ldb == 256)
      gemm_f32_wmma<0, 256><<<grid, 128, 0, stream>>>(A, Bm, C, K, sA, sB, sC, lda, ldc,
                                                      bias, alpha, beta);
    else if (ldb == 512)
      gemm_f32_wmma<0, 512><<<grid, 128, 0, stream>>>(A, Bm, C, K, sA, sB, sC, lda, ldc,
                                                      bias, alpha, beta);
    else
      gemm_f32_wmma<0, 1536><<<grid, 128, 0, stream>>>(A, Bm, C, K, sA, sB, sC, lda, ldc,
                                                       bias, alpha, beta);
  };
  auto conv = [&](const _Float16* Xc, const float* W, _Float16* Y, int CI, int CO,
                  int KW, int Ctot, int coOff, const float* g, const float* b,
                  int relu) {
    dim3 grid(64, CO / 128, 8);
    switch (KW) {
      case 1:
        conv1d_wmma<1><<<grid, 128, 0, stream>>>(Xc, W, Y, CI, Ctot, coOff, g, b, relu);
        break;
      case 9:
        conv1d_wmma<9><<<grid, 128, 0, stream>>>(Xc, W, Y, CI, Ctot, coOff, g, b, relu);
        break;
      case 19:
        conv1d_wmma<19><<<grid, 128, 0, stream>>>(Xc, W, Y, CI, Ctot, coOff, g, b, relu);
        break;
      default:
        conv1d_wmma<39><<<grid, 128, 0, stream>>>(Xc, W, Y, CI, Ctot, coOff, g, b, relu);
        break;
    }
  };

  // ---------------- Phase A: InceptionTime ----------------
  long inTot = (long)8 * 32 * 1024;
  k_prep_input<<<ew(inTot), 256, 0, stream>>>(X, Hp(O_XA), Hp(O_RES), inTot);
  _Float16* cur = Hp(O_XA);
  _Float16* nxt = Hp(O_XB);
  int curCI = 32;
  const int ks[3] = {39, 19, 9};
  for (int d = 0; d < 6; ++d) {
    int base = 1 + 7 * d;
    conv(cur, P(base + 0), Hp(O_XBOT), curCI, 128, 1, 128, 0, nullptr, nullptr, 0);
    for (int i = 0; i < 3; ++i)
      conv(Hp(O_XBOT), P(base + 1 + i), nxt, 128, 128, ks[i], 512, i * 128,
           P(base + 5), P(base + 6), 1);
    long mpTot = (long)8 * curCI * 1024;
    k_maxpool3<<<ew(mpTot), 256, 0, stream>>>(cur, Hp(O_MP), mpTot, 1024);
    conv(Hp(O_MP), P(base + 4), nxt, curCI, 128, 1, 512, 384, P(base + 5), P(base + 6), 1);
    long hTot = (long)8 * 512 * 1024;
    if (d == 2) {
      conv(Hp(O_RES), P(43), Hp(O_S), 32, 512, 1, 512, 0, P(44), P(45), 0);
      k_res_add_relu<<<ew(hTot), 256, 0, stream>>>(nxt, Hp(O_S), (const _Float16*)nullptr,
                                                   (const float*)nullptr,
                                                   (const float*)nullptr, 0, Hp(O_RES),
                                                   hTot);
    } else if (d == 5) {
      k_res_add_relu<<<ew(hTot), 256, 0, stream>>>(nxt, (const _Float16*)nullptr,
                                                   Hp(O_RES), P(46), P(47), 1, Hp(O_RES),
                                                   hTot);
    }
    _Float16* t = cur; cur = nxt; nxt = t;
    curCI = 512;
  }
  long tokTot = (long)8 * 1026 * 512;
  k_build_tokens<<<ew(tokTot), 256, 0, stream>>>(cur, P(48), F(O_TOK), tokTot);

  // ---------------- Phase B helpers ----------------
  auto wavelet = [&](const float* w1, const float* w2, const float* w3,
                     const float* pw, const float* pb) {
    long n = (long)8 * 1024 * 512;
    k_wavelet_pos<<<ew(n), 256, 0, stream>>>(F(O_TOK), w1, w2, w3, F(O_POS), n);
    // feat += pos @ pe_w + pe_b  (batched over B with 1026-row token stride)
    gemm(F(O_POS), pw, F(O_TOK) + 2 * 512, 1024, 512, 512, (long)1024 * 512, 0,
         (long)1026 * 512, 512, 512, 512, 0, pb, 1.f, 1.f, 8);
  };

  auto attention = [&](const float* qkv_w, const float* out_w, const float* out_b,
                       const float* res_w, const float* ln_g, const float* ln_b) {
    long npadTot = (long)8 * 1280 * 512;
    long sqTot = (long)64 * 256 * 256;
    long sS = (long)256 * 256;
    k_layernorm<<<8 * 1026, 256, 0, stream>>>(F(O_TOK), ln_g, ln_b, F(O_LNX));
    k_pad_tokens<<<ew(npadTot), 256, 0, stream>>>(F(O_LNX), F(O_XPAD), npadTot);
    gemm(F(O_XPAD), qkv_w, F(O_QKVRAW), 8 * 1280, 1536, 512, 0, 0, 0, 512, 1536, 1536,
         0, nullptr, 1.f, 0.f, 1);
    k_split_qkv<<<ew(npadTot), 256, 0, stream>>>(F(O_QKVRAW), F(O_Q), F(O_K), F(O_V),
                                                 npadTot);
    long lmTot = (long)64 * 256 * 64;
    k_landmarks<<<ew(lmTot), 256, 0, stream>>>(F(O_Q), F(O_K), F(O_QL), F(O_KL), lmTot);
    // a1 = softmax(q @ kl^T), a2 = softmax(ql @ kl^T), a3 = softmax(ql @ k^T)
    gemm(F(O_Q), F(O_KL), F(O_P0), 1280, 256, 64, (long)1280 * 64, (long)256 * 64,
         (long)1280 * 256, 64, 64, 256, 1, nullptr, 1.f, 0.f, 64);
    k_softmax_rows<<<64 * 1280, 256, 0, stream>>>(F(O_P0), 256);
    gemm(F(O_QL), F(O_KL), F(O_A2), 256, 256, 64, (long)256 * 64, (long)256 * 64, sS,
         64, 64, 256, 1, nullptr, 1.f, 0.f, 64);
    k_softmax_rows<<<64 * 256, 256, 0, stream>>>(F(O_A2), 256);
    gemm(F(O_QL), F(O_K), F(O_A3), 256, 1280, 64, (long)256 * 64, (long)1280 * 64,
         (long)256 * 1280, 64, 64, 1280, 1, nullptr, 1.f, 0.f, 64);
    k_softmax_rows<<<64 * 256, 256, 0, stream>>>(F(O_A3), 1280);
    // a3 @ v  (before AOUT reuses the P1 arena)
    gemm(F(O_A3), F(O_V), F(O_A3V), 256, 64, 1280, (long)256 * 1280, (long)1280 * 64,
         (long)256 * 64, 1280, 64, 64, 0, nullptr, 1.f, 0.f, 64);
    // Moore-Penrose pinv of a2 (6 Newton iterations, global denom)
    k_pinv_sums<<<64, 256, 0, stream>>>(F(O_A2), F(O_RC));
    k_pinv_denom<<<1, 64, 0, stream>>>(F(O_RC), F(O_DEN));
    k_pinv_init<<<ew(sqTot), 256, 0, stream>>>(F(O_A2), F(O_DEN), F(O_Z), sqTot);
    float* zp = F(O_Z); float* zq = F(O_ZB); float* t1 = F(O_T1); float* t2 = F(O_T2);
    for (int it = 0; it < 6; ++it) {
      gemm(F(O_A2), zp, zq, 256, 256, 256, sS, sS, sS, 256, 256, 256, 0, nullptr, 1.f,
           0.f, 64);                                                   // xz
      k_id_sub<<<ew(sqTot), 256, 0, stream>>>(zq, t1, 7.f, sqTot);     // 7I - xz
      gemm(zq, t1, t2, 256, 256, 256, sS, sS, sS, 256, 256, 256, 0, nullptr, 1.f, 0.f,
           64);                                                        // xz@(...)
      k_id_sub<<<ew(sqTot), 256, 0, stream>>>(t2, t2, 15.f, sqTot);    // 15I - ...
      gemm(zq, t2, t1, 256, 256, 256, sS, sS, sS, 256, 256, 256, 0, nullptr, 1.f, 0.f,
           64);                                                        // xz@(...)
      k_id_sub<<<ew(sqTot), 256, 0, stream>>>(t1, t1, 13.f, sqTot);    // 13I - ...
      gemm(zp, t1, zq, 256, 256, 256, sS, sS, sS, 256, 256, 256, 0, nullptr, 0.25f, 0.f,
           64);                                                        // z = .25 z@(...)
      float* tmp = zp; zp = zq; zq = tmp;
    }
    // out = a1 @ z @ (a3@v) + depthwise(v)
    gemm(F(O_P0), zp, F(O_AP), 1280, 256, 256, (long)1280 * 256, sS, (long)1280 * 256,
         256, 256, 256, 0, nullptr, 1.f, 0.f, 64);
    gemm(F(O_AP), F(O_A3V), F(O_AOUT), 1280, 64, 256, (long)1280 * 256, (long)256 * 64,
         (long)1280 * 64, 256, 64, 64, 0, nullptr, 1.f, 0.f, 64);
    long vTot = (long)64 * 1280 * 64;
    k_resconv<<<ew(vTot), 256, 0, stream>>>(F(O_AOUT), F(O_V), res_w, vTot);
    k_merge<<<ew(npadTot), 256, 0, stream>>>(F(O_AOUT), F(O_MERG), npadTot);
    gemm(F(O_MERG), out_w, F(O_PROJ), 8 * 1280, 512, 512, 0, 0, 0, 512, 512, 512, 0,
         out_b, 1.f, 0.f, 1);
    k_add_attn<<<ew((long)8 * 1026 * 512), 256, 0, stream>>>(F(O_TOK), F(O_PROJ),
                                                             (long)8 * 1026 * 512);
  };

  // ---------------- Phase B: transformer ----------------
  wavelet(P(49), P(50), P(51), P(55), P(56));
  attention(P(63), P(64), P(65), P(66), P(59), P(60));
  wavelet(P(52), P(53), P(54), P(57), P(58));
  attention(P(67), P(68), P(69), P(70), P(61), P(62));

  k_final<<<16, 256, 0, stream>>>(F(O_TOK), (float*)d_out);
}
